// Encoder_26852135534742
// MI455X (gfx1250) — compile-verified, matrix operand order checked by reading, not run on hardware
//
#include <hip/hip_runtime.h>
#include <hip/hip_bf16.h>
#include <stdint.h>

// ---------------------------------------------------------------------------
// Encoder (attention-LSTM), B=512 T-1=128 D=128 H=512, for gfx1250 (MI455X).
//
// softmax(proj_x + shift[:,None]) == softmax(proj_x)  (shift-invariant), so
// attention weights are time-invariant; wi[b,t,:] is precomputed.
//
// Recurrence: gates = [wi_t | h] @ [W_ih ; W_hh]^T + (b_ih+b_hh), K=640,
// v_wmma_f32_16x16x32_bf16 (fp32 accumulate), 20 K-tiles. Each wave owns
// TWO 16-row batch tiles x all four gates (8 accumulators): every B tile
// feeds 2 WMMAs, every A tile feeds 4, and K-tiles are explicitly
// double-buffered so loads for kt+1 overlap the 8 WMMAs of kt.
// ---------------------------------------------------------------------------

typedef __attribute__((ext_vector_type(16))) __bf16 v16bf;
typedef __attribute__((ext_vector_type(8)))  float  v8f;

union BfPack { uint4 q[2]; v16bf v; };

// round-to-nearest-even float -> bf16 (raw bits)
__device__ __forceinline__ unsigned short f2bf(float x) {
    unsigned int u = __float_as_uint(x);
    unsigned int r = u + 0x7FFFu + ((u >> 16) & 1u);
    return (unsigned short)(r >> 16);
}

// Load 16 bf16 as two 16-byte chunks (ISA 7.12.2 16-bit A/B lane layout).
__device__ __forceinline__ v16bf load16(const unsigned short* lo,
                                        const unsigned short* hi) {
    BfPack p;
    p.q[0] = *(const uint4*)lo;
    p.q[1] = *(const uint4*)hi;
    return p.v;
}

__device__ __forceinline__ float sigf(float x) {
    return 1.0f / (1.0f + __expf(-x));
}

#define WMMA_BF16(A, B, C) \
    __builtin_amdgcn_wmma_f32_16x16x32_bf16(false, (A), false, (B), (short)0, (C), false, false)

// ---------------------------------------------------------------------------
// Kernel A: proj_x -> softmax (time-invariant) -> wi for all t.
// ---------------------------------------------------------------------------
__global__ void __launch_bounds__(128)
attn_wi_kernel(const float* __restrict__ x, const float* __restrict__ attn_W,
               float* __restrict__ out_w, unsigned short* __restrict__ wi_bf) {
    const int b = blockIdx.x;      // 0..511
    const int d = threadIdx.x;     // 0..127
    const float* xb = x + (size_t)b * 128 * 128;
    const float* wx = attn_W + 1024;   // attn_W[0][2H:] = wx[t]

    float s = 0.0f;
    for (int t = 0; t < 128; ++t) s = fmaf(xb[t * 128 + d], wx[t], s);

    __shared__ float red[128];
    red[d] = s; __syncthreads();
    for (int off = 64; off > 0; off >>= 1) {
        if (d < off) red[d] = fmaxf(red[d], red[d + off]);
        __syncthreads();
    }
    const float mx = red[0]; __syncthreads();
    const float e = __expf(s - mx);
    red[d] = e; __syncthreads();
    for (int off = 64; off > 0; off >>= 1) {
        if (d < off) red[d] += red[d + off];
        __syncthreads();
    }
    const float attn = e / red[0];

    for (int t = 0; t < 128; ++t) {
        const float wi = attn * xb[t * 128 + d];
        const size_t o = (size_t)(b * 128 + t) * 128 + d;
        out_w[o] = wi;
        wi_bf[o] = f2bf(wi);
    }
}

// ---------------------------------------------------------------------------
// Kernel B: Wcat[n][k] (bf16, k<128 -> W_ih, else W_hh), bias = b_ih + b_hh.
// ---------------------------------------------------------------------------
__global__ void __launch_bounds__(256)
prep_kernel(const float* __restrict__ W_ih, const float* __restrict__ W_hh,
            const float* __restrict__ b_ih, const float* __restrict__ b_hh,
            unsigned short* __restrict__ Wcat, float* __restrict__ bias) {
    const int idx = blockIdx.x * blockDim.x + threadIdx.x;
    if (idx < 2048 * 640) {
        const int n = idx / 640;
        const int k = idx - n * 640;
        const float w = (k < 128) ? W_ih[n * 128 + k]
                                  : W_hh[n * 512 + (k - 128)];
        Wcat[idx] = f2bf(w);
    }
    if (idx < 2048) bias[idx] = b_ih[idx] + b_hh[idx];
}

// ---------------------------------------------------------------------------
// Kernel C: one LSTM step. grid = (16 batch tiles of 32, 8 j-groups),
// 128 threads = 4 waves; wave w owns j0 = by*64 + w*16 for 2 batch tiles
// x 4 gates = 8 accumulators, 160 WMMAs per wave per step.
// ---------------------------------------------------------------------------
__global__ void __launch_bounds__(128)
lstm_step_kernel(const unsigned short* __restrict__ wi_bf,
                 const unsigned short* __restrict__ Wcat,
                 const float* __restrict__ bias,
                 const unsigned short* __restrict__ h_prev,
                 unsigned short* __restrict__ h_next,
                 float* __restrict__ c_state,
                 float* __restrict__ out_enc, int t) {
    const int lane  = threadIdx.x & 31;
    const int wave  = threadIdx.x >> 5;              // j sub-tile 0..3
    const int bx    = blockIdx.x;                    // batch tile 0..15 (x32)
    const int j0    = blockIdx.y * 64 + wave * 16;   // hidden column tile base
    const int mrow  = lane & 15;                     // tile row (A) / col (B,D)
    const int khalf = (lane >> 4) * 8;               // K sub-chunk (0 or 8)

    // A-operand rows for the two batch sub-tiles.
    const int bA0 = bx * 32 + mrow;
    const int bA1 = bA0 + 16;
    const unsigned short* wiRow0 = wi_bf + ((size_t)(bA0 * 128 + t)) * 128;
    const unsigned short* wiRow1 = wi_bf + ((size_t)(bA1 * 128 + t)) * 128;
    const unsigned short* hRow0  = h_prev + (size_t)bA0 * 512;
    const unsigned short* hRow1  = h_prev + (size_t)bA1 * 512;

    // B-operand rows: gate g output neuron n = g*512 + j0 + mrow.
    const unsigned short* wRow0 = Wcat + (size_t)(0 * 512 + j0 + mrow) * 640;
    const unsigned short* wRow1 = Wcat + (size_t)(1 * 512 + j0 + mrow) * 640;
    const unsigned short* wRow2 = Wcat + (size_t)(2 * 512 + j0 + mrow) * 640;
    const unsigned short* wRow3 = Wcat + (size_t)(3 * 512 + j0 + mrow) * 640;

    auto aPtr0 = [&](int kt) {
        return (kt < 4) ? (wiRow0 + kt * 32) : (hRow0 + kt * 32 - 128);
    };
    auto aPtr1 = [&](int kt) {
        return (kt < 4) ? (wiRow1 + kt * 32) : (hRow1 + kt * 32 - 128);
    };

    v8f aI0 = {}, aF0 = {}, aG0 = {}, aO0 = {};   // batch sub-tile 0
    v8f aI1 = {}, aF1 = {}, aG1 = {}, aO1 = {};   // batch sub-tile 1

    // Prime the pipeline: K-tile 0.
    const unsigned short* p0 = aPtr0(0);
    const unsigned short* p1 = aPtr1(0);
    v16bf a0 = load16(p0 + khalf, p0 + 16 + khalf);
    v16bf a1 = load16(p1 + khalf, p1 + 16 + khalf);
    v16bf b0 = load16(wRow0 + khalf, wRow0 + 16 + khalf);
    v16bf b1 = load16(wRow1 + khalf, wRow1 + 16 + khalf);
    v16bf b2 = load16(wRow2 + khalf, wRow2 + 16 + khalf);
    v16bf b3 = load16(wRow3 + khalf, wRow3 + 16 + khalf);

#pragma unroll
    for (int kt = 0; kt < 20; ++kt) {
        v16bf a0n, a1n, b0n, b1n, b2n, b3n;
        if (kt < 19) {   // prefetch K-tile kt+1 into fresh registers
            const int k0 = (kt + 1) * 32;
            const unsigned short* q0 = aPtr0(kt + 1);
            const unsigned short* q1 = aPtr1(kt + 1);
            a0n = load16(q0 + khalf,            q0 + 16 + khalf);
            a1n = load16(q1 + khalf,            q1 + 16 + khalf);
            b0n = load16(wRow0 + k0 + khalf,    wRow0 + k0 + 16 + khalf);
            b1n = load16(wRow1 + k0 + khalf,    wRow1 + k0 + 16 + khalf);
            b2n = load16(wRow2 + k0 + khalf,    wRow2 + k0 + 16 + khalf);
            b3n = load16(wRow3 + k0 + khalf,    wRow3 + k0 + 16 + khalf);
        }
        aI0 = WMMA_BF16(a0, b0, aI0);
        aI1 = WMMA_BF16(a1, b0, aI1);
        aF0 = WMMA_BF16(a0, b1, aF0);
        aF1 = WMMA_BF16(a1, b1, aF1);
        aG0 = WMMA_BF16(a0, b2, aG0);
        aG1 = WMMA_BF16(a1, b2, aG1);
        aO0 = WMMA_BF16(a0, b3, aO0);
        aO1 = WMMA_BF16(a1, b3, aO1);
        if (kt < 19) {
            a0 = a0n; a1 = a1n;
            b0 = b0n; b1 = b1n; b2 = b2n; b3 = b3n;
        }
    }

    // This lane's output column and per-gate biases.
    const int j = j0 + mrow;
    const float bi = bias[0 * 512 + j];
    const float bf = bias[1 * 512 + j];
    const float bg = bias[2 * 512 + j];
    const float bo = bias[3 * 512 + j];

#pragma unroll
    for (int r = 0; r < 8; ++r) {
        const int m = r + (lane >> 4) * 8;           // D-matrix row layout
        // ---- batch sub-tile 0 ----
        {
            const int b = bx * 32 + m;
            const size_t cidx = (size_t)b * 512 + j;
            const float c_old = c_state[cidx];
            const float cn = sigf(aF0[r] + bf) * c_old
                           + sigf(aI0[r] + bi) * tanhf(aG0[r] + bg);
            const float hn = sigf(aO0[r] + bo) * tanhf(cn);
            c_state[cidx] = cn;
            h_next[cidx]  = f2bf(hn);
            out_enc[((size_t)(b * 128 + t)) * 512 + j] = hn;
        }
        // ---- batch sub-tile 1 ----
        {
            const int b = bx * 32 + 16 + m;
            const size_t cidx = (size_t)b * 512 + j;
            const float c_old = c_state[cidx];
            const float cn = sigf(aF1[r] + bf) * c_old
                           + sigf(aI1[r] + bi) * tanhf(aG1[r] + bg);
            const float hn = sigf(aO1[r] + bo) * tanhf(cn);
            c_state[cidx] = cn;
            h_next[cidx]  = f2bf(hn);
            out_enc[((size_t)(b * 128 + t)) * 512 + j] = hn;
        }
    }
}

// ---------------------------------------------------------------------------
extern "C" void kernel_launch(void* const* d_in, const int* in_sizes, int n_in,
                              void* d_out, int out_size, void* d_ws, size_t ws_size,
                              hipStream_t stream) {
    const float* x      = (const float*)d_in[0];   // (512,128,128)
    const float* W_ih   = (const float*)d_in[1];   // (2048,128)
    const float* W_hh   = (const float*)d_in[2];   // (2048,512)
    const float* b_ih   = (const float*)d_in[3];   // (2048,)
    const float* b_hh   = (const float*)d_in[4];   // (2048,)
    const float* attn_W = (const float*)d_in[5];   // (1,1152)
    (void)d_in[6]; (void)in_sizes; (void)n_in; (void)out_size; (void)ws_size;
    // attn_b unused: softmax is invariant to the per-row shift.

    float* out_w   = (float*)d_out;                              // (512,128,128)
    float* out_enc = (float*)d_out + (size_t)512 * 128 * 128;    // (512,128,512)

    char* ws = (char*)d_ws;
    unsigned short* wi_bf  = (unsigned short*)ws; ws += (size_t)512 * 128 * 128 * 2; // 16 MB
    unsigned short* Wcat   = (unsigned short*)ws; ws += (size_t)2048 * 640 * 2;      // 2.6 MB
    float*          bias   = (float*)ws;          ws += (size_t)2048 * 4;
    unsigned short* h0     = (unsigned short*)ws; ws += (size_t)512 * 512 * 2;
    unsigned short* h1     = (unsigned short*)ws; ws += (size_t)512 * 512 * 2;
    float*          cstate = (float*)ws;          ws += (size_t)512 * 512 * 4;

    hipMemsetAsync(h0,     0, (size_t)512 * 512 * 2, stream);
    hipMemsetAsync(cstate, 0, (size_t)512 * 512 * 4, stream);

    attn_wi_kernel<<<512, 128, 0, stream>>>(x, attn_W, out_w, wi_bf);
    prep_kernel<<<(2048 * 640 + 255) / 256, 256, 0, stream>>>(
        W_ih, W_hh, b_ih, b_hh, Wcat, bias);

    for (int t = 0; t < 128; ++t) {
        const unsigned short* hp = (t & 1) ? h1 : h0;
        unsigned short*       hn = (t & 1) ? h0 : h1;
        lstm_step_kernel<<<dim3(16, 8), 128, 0, stream>>>(
            wi_bf, Wcat, bias, hp, hn, cstate, out_enc, t);
    }
}
